// MultiHeadAttention_23046794510725
// MI455X (gfx1250) — matrix operand order repaired and checked by reading, not compiled, not run on hardware
//
#include <hip/hip_runtime.h>

// ---------------------------------------------------------------------------
// MI455X (gfx1250, wave32) fused multi-head attention, bf16 WMMA everywhere,
// async-to-LDS / TDM tile staging, double-buffered GEMM.
//   B=4, T=2048, D=1024, H=16, HD=64
// ---------------------------------------------------------------------------

#define B_   4
#define T_   2048
#define D_   1024
#define H_   16
#define HD_  64
#define M_   (B_ * T_)       // 8192 rows for all projections

typedef __attribute__((ext_vector_type(16))) __bf16 v16bf;
typedef __attribute__((ext_vector_type(8)))  __bf16 v8bf;
typedef __attribute__((ext_vector_type(8)))  float  v8f;
typedef __attribute__((ext_vector_type(4)))  __bf16 v4bf;
typedef __attribute__((ext_vector_type(4)))  float  v4f;
typedef __attribute__((ext_vector_type(4)))  unsigned int u32x4;
typedef __attribute__((ext_vector_type(8)))  int    i32x8;
typedef __attribute__((ext_vector_type(4)))  int    i32x4;

#ifndef __has_builtin
#define __has_builtin(x) 0
#endif

#if __has_builtin(__builtin_amdgcn_global_load_async_to_lds_b128) && \
    __has_builtin(__builtin_amdgcn_s_wait_asynccnt)
#define USE_GASYNC 1
#else
#define USE_GASYNC 0
#endif

#if __has_builtin(__builtin_amdgcn_tensor_load_to_lds) && \
    __has_builtin(__builtin_amdgcn_s_wait_tensorcnt)
#define USE_TDM 1
#else
#define USE_TDM 0
#endif

// Exact parameter types for the async-to-LDS builtin (from hipcc diagnostic:
// param 0 is 'vector_size(16) int __device__ *', i.e. v4i in AS1, non-const).
typedef int v4i_vs __attribute__((__vector_size__(4 * sizeof(int))));
typedef __attribute__((address_space(1))) v4i_vs gas1_v4i;  // global
typedef __attribute__((address_space(3))) v4i_vs las3_v4i;  // LDS

// 16-byte global -> LDS copy: async (bypasses VGPRs, ASYNCcnt) when available.
__device__ __forceinline__ void cp16(void* dst_lds, const void* src_gbl) {
#if USE_GASYNC
  __builtin_amdgcn_global_load_async_to_lds_b128(
      (gas1_v4i*)(uintptr_t)src_gbl,
      (las3_v4i*)(unsigned)(uintptr_t)dst_lds, 0, 0);
#else
  *(v8bf*)dst_lds = *(const v8bf*)src_gbl;
#endif
}

// Join: make this wave's async LDS fills complete, then block barrier.
__device__ __forceinline__ void cp_join() {
#if USE_GASYNC
  __builtin_amdgcn_s_wait_asynccnt(0);
#endif
  __syncthreads();
}

#if USE_TDM
// Tensor Data Mover: 2D tile (64 x 64 bf16) from global (row pitch
// `row_stride` elements) into contiguous LDS. D# built per CDNA5 ISA §8.3/8.4.
__device__ __forceinline__ void tdm_load_2d_64x64(void* lds, const void* gsrc,
                                                  unsigned row_stride) {
  const unsigned tile_d0 = 64, tile_d1 = 64;     // elements (bf16)
  unsigned lds_addr = (unsigned)(uintptr_t)lds;  // low 32 bits = LDS offset
  unsigned long long ga = (unsigned long long)(uintptr_t)gsrc;
  u32x4 g0 = {};
  g0[0] = 1u;                                    // count=1, user descriptor
  g0[1] = lds_addr;                              // lds_addr [63:32]
  g0[2] = (unsigned)(ga & 0xffffffffu);          // global_addr[31:0]
  g0[3] = (unsigned)((ga >> 32) & 0x1ffffffu) |  // global_addr[56:32]
          (2u << 30);                            // type=2 ("image")
  i32x8 g1 = {};
  g1[0] = (int)(1u << 16);                       // wg_mask=0, data_size=1 (2B)
  g1[1] = (int)((tile_d0 & 0xffffu) << 16);      // tensor_dim0[15:0] @ [63:48]
  g1[2] = (int)((tile_d0 >> 16) | ((tile_d1 & 0xffffu) << 16)); // d0 hi | d1 lo
  g1[3] = (int)((tile_d1 >> 16) | (tile_d0 << 16));             // d1 hi | tile_dim0
  g1[4] = (int)(tile_d1 & 0xffffu);              // tile_dim1 (tile_dim2=0)
  g1[5] = (int)row_stride;                       // tensor_dim0_stride[31:0]
  g1[6] = 0;                                     // stride hi, dim1_stride lo
  g1[7] = 0;
  i32x4 gz = {};
#if defined(__clang_major__) && __clang_major__ >= 23
  i32x8 gz8 = {};
  __builtin_amdgcn_tensor_load_to_lds(g0, g1, gz, gz, gz8, 0);
#else
  __builtin_amdgcn_tensor_load_to_lds(g0, g1, gz, gz, 0);
#endif
}
#endif

// ---- WMMA fragment builders (per CDNA5 ISA 16-bit layouts, wave32) --------
__device__ __forceinline__ v16bf frag_a(const __bf16* __restrict__ p, int hi) {
  v16bf f;
#pragma unroll
  for (int i = 0; i < 8; ++i) {
    int kb = 2 * (i & 3) + ((i & 4) ? 16 : 0) + hi * 8;
    f[2 * i]     = p[kb];
    f[2 * i + 1] = p[kb + 1];
  }
  return f;
}
__device__ __forceinline__ v16bf frag_b(const __bf16* __restrict__ p, int hi) {
  v16bf f;
#pragma unroll
  for (int i = 0; i < 8; ++i) {
    int kb = 2 * (i & 3) + ((i & 4) ? 16 : 0) + hi * 8;
    f[2 * i]     = p[kb];
    f[2 * i + 1] = p[kb + 1];
  }
  return f;
}

// ---------------------------------------------------------------------------
// fp32 -> bf16 conversion (4-wide, linear)
// ---------------------------------------------------------------------------
__global__ void f32_to_bf16_kernel(const float* __restrict__ src,
                                   __bf16* __restrict__ dst, int n4) {
  int i = blockIdx.x * blockDim.x + threadIdx.x;
  if (i < n4) {
    v4f f = *(const v4f*)&src[i * 4];
    v4bf o;
#pragma unroll
    for (int j = 0; j < 4; ++j) o[j] = (__bf16)f[j];
    *(v4bf*)&dst[i * 4] = o;
  }
}

// ---------------------------------------------------------------------------
// fp32 [K][N] -> bf16 [N][K] transposed conversion (weights): removes the
// LDS-side transpose from the GEMM so every tile fill is a straight 2D copy.
// ---------------------------------------------------------------------------
__global__ void transpose_f32_to_bf16_kernel(const float* __restrict__ src,
                                             __bf16* __restrict__ dst,
                                             int K, int N) {
  __shared__ float t[32][33];
  int n0 = blockIdx.x * 32, k0 = blockIdx.y * 32;
#pragma unroll
  for (int i = 0; i < 4; ++i) {
    int r = threadIdx.y + i * 8;
    t[r][threadIdx.x] = src[(size_t)(k0 + r) * N + n0 + threadIdx.x];
  }
  __syncthreads();
#pragma unroll
  for (int i = 0; i < 4; ++i) {
    int r = threadIdx.y + i * 8;
    dst[(size_t)(n0 + r) * K + k0 + threadIdx.x] = (__bf16)t[threadIdx.x][r];
  }
}

// ---------------------------------------------------------------------------
// bf16 GEMM: C[MxN] = A[MxK] * Bt[NxK]^T   (Bt pre-transposed, row-major NxK)
// Block tile 64x128, 8 waves, wave 32x32 (2x2 WMMA), BK=32,
// double-buffered LDS with async global->LDS fills (1 barrier / K-step).
// MODE: 0 = bf16 out, 1 = bf16 out in V-transposed [b][h][hd][T] layout,
//       2 = f32 out + bias.
// ---------------------------------------------------------------------------
template <int MODE>
__global__ __launch_bounds__(256) void gemm_bf16_kernel(
    const __bf16* __restrict__ A, const __bf16* __restrict__ Bt,
    __bf16* __restrict__ Cb, float* __restrict__ Cf,
    const float* __restrict__ bias, int M, int N, int K) {
  __shared__ __bf16 As[2][64][32];    // [m][k]
  __shared__ __bf16 Bs[2][128][32];   // [n][k]

  const int tid  = threadIdx.x;
  const int wave = tid >> 5;
  const int lane = tid & 31;
  const int hi   = lane >> 4;
  const int ln   = lane & 15;
  const int wr   = wave >> 2;
  const int wc   = wave & 3;
  const int rowBase = blockIdx.y * 64;
  const int colBase = blockIdx.x * 128;

  v8f acc[2][2] = {};

  auto load_tiles = [&](int bsel, int kk) {
    {  // A tile 64x32: one b128 per thread
      int r = tid >> 2, k8 = (tid & 3) * 8;
      cp16(&As[bsel][r][k8], &A[(size_t)(rowBase + r) * K + kk + k8]);
    }
#pragma unroll
    for (int it = 0; it < 2; ++it) {  // B tile 128x32: two b128 per thread
      int c = tid + it * 256;
      int n = c >> 2, k8 = (c & 3) * 8;
      cp16(&Bs[bsel][n][k8], &Bt[(size_t)(colBase + n) * K + kk + k8]);
    }
  };

  load_tiles(0, 0);
  int buf = 0;
  for (int kk = 0; kk < K; kk += 32) {
    cp_join();  // tiles[buf] ready; everyone done reading tiles[buf^1]
    if (kk + 32 < K) load_tiles(buf ^ 1, kk + 32);  // prefetch next (async)

    v16bf af[2], bfr[2];
#pragma unroll
    for (int sm = 0; sm < 2; ++sm)
      af[sm] = frag_a(&As[buf][wr * 32 + sm * 16 + ln][0], hi);
#pragma unroll
    for (int sn = 0; sn < 2; ++sn)
      bfr[sn] = frag_b(&Bs[buf][wc * 32 + sn * 16 + ln][0], hi);
#pragma unroll
    for (int sm = 0; sm < 2; ++sm)
#pragma unroll
      for (int sn = 0; sn < 2; ++sn)
        acc[sm][sn] = __builtin_amdgcn_wmma_f32_16x16x32_bf16(
            false, af[sm], false, bfr[sn], (short)0, acc[sm][sn], false, false);
    buf ^= 1;
  }

  // ---- epilogue: C/D layout row = r + 8*hi, col = lane&15 ----
#pragma unroll
  for (int sm = 0; sm < 2; ++sm)
#pragma unroll
    for (int sn = 0; sn < 2; ++sn)
#pragma unroll
      for (int r = 0; r < 8; ++r) {
        int row = rowBase + wr * 32 + sm * 16 + r + hi * 8;
        int col = colBase + wc * 32 + sn * 16 + ln;
        float v = acc[sm][sn][r];
        if constexpr (MODE == 2) {
          Cf[(size_t)row * N + col] = v + bias[col];
        } else if constexpr (MODE == 1) {
          // V stored [b][h][hd][T] so attention PV tiles are direct 2D copies
          int bb = row >> 11, t = row & (T_ - 1);
          int hh = col >> 6, hd = col & (HD_ - 1);
          Cb[((size_t)((bb * H_ + hh) * HD_ + hd)) * T_ + t] = (__bf16)v;
        } else {
          Cb[(size_t)row * N + col] = (__bf16)v;
        }
      }
}

// ---------------------------------------------------------------------------
// Fused causal flash attention. One block = one (b,h) x one 64-query tile.
// 8 waves; wave w: row group w>>1, column tiles {2*(w&1), 2*(w&1)+1}.
// K tile [key][hd] and V tile [hd][key] (pre-transposed layout) staged via
// TDM (TENSORcnt) when available, else async-to-LDS; online softmax.
// ---------------------------------------------------------------------------
__global__ __launch_bounds__(256) void attn_kernel(
    const __bf16* __restrict__ Qg, const __bf16* __restrict__ Kg,
    const __bf16* __restrict__ Vtg, __bf16* __restrict__ Og) {
  __shared__ __bf16 Qs[64][64];   // [q][hd]
  __shared__ __bf16 Ks[64][64];   // [key][hd]
  __shared__ __bf16 Vs[64][64];   // [hd][key]
  __shared__ float  Ss[64][64];
  __shared__ __bf16 Ps[64][64];
  __shared__ float  pmax[64][4], psum[64][4];
  __shared__ float  mrow[64], lrow[64], arow[64], mbc[64];

  const int tid  = threadIdx.x;
  const int wave = tid >> 5;
  const int lane = tid & 31;
  const int hi   = lane >> 4;
  const int ln   = lane & 15;
  const int rg   = wave >> 1;
  const int ct0  = (wave & 1) * 2;

  const int bid = blockIdx.x;
  const int qt  = bid & 31;
  const int bh  = bid >> 5;
  const int b   = bh >> 4;
  const int h   = bh & 15;
  const int q0  = qt * 64;

  const __bf16* Qbase  = Qg  + ((size_t)(b * T_ + q0)) * D_ + h * HD_;
  const __bf16* Vtbase = Vtg + ((size_t)(b * H_ + h)) * HD_ * T_;

#pragma unroll
  for (int it = 0; it < 2; ++it) {  // Q tile 64x64: two b128 per thread
    int c = tid + it * 256;
    int r = c >> 3, c8 = (c & 7) * 8;
    cp16(&Qs[r][c8], &Qbase[(size_t)r * D_ + c8]);
  }
  if (tid < 64) { mrow[tid] = -3.0e38f; lrow[tid] = 0.0f; }

  v8f oacc[2] = {};
  const int prow = tid >> 2, pseg = tid & 3;  // softmax: 4 x 16-col segments

  for (int kc = 0; kc <= qt; ++kc) {
    const __bf16* Kbase = Kg + ((size_t)(b * T_ + kc * 64)) * D_ + h * HD_;
    const __bf16* Vbase = Vtbase + kc * 64;   // rows = hd, pitch = T_
#if USE_TDM
    if (wave == 0) {
      tdm_load_2d_64x64(&Ks[0][0], Kbase, D_);
      tdm_load_2d_64x64(&Vs[0][0], Vbase, T_);
      __builtin_amdgcn_s_wait_tensorcnt(0);
    }
#else
#pragma unroll
    for (int it = 0; it < 2; ++it) {
      int c = tid + it * 256;
      int r = c >> 3, c8 = (c & 7) * 8;
      cp16(&Ks[r][c8], &Kbase[(size_t)r * D_ + c8]);
      cp16(&Vs[r][c8], &Vbase[(size_t)r * T_ + c8]);
    }
#endif
    cp_join();  // [A] K/V (and first-iter Q) tiles visible to all waves

    // ---- S = (Q K^T) * rsqrt(HD), causal-masked ----
#pragma unroll
    for (int t = 0; t < 2; ++t) {
      int ctile = ct0 + t;
      v8f sacc = {};
#pragma unroll
      for (int step = 0; step < 2; ++step) {
        v16bf a  = frag_a(&Qs[rg * 16 + ln][step * 32], hi);
        v16bf bb = frag_b(&Ks[ctile * 16 + ln][step * 32], hi);
        sacc = __builtin_amdgcn_wmma_f32_16x16x32_bf16(
            false, a, false, bb, (short)0, sacc, false, false);
      }
#pragma unroll
      for (int r = 0; r < 8; ++r) {
        int row = rg * 16 + r + hi * 8;
        int col = ctile * 16 + ln;
        float s = sacc[r] * 0.125f;               // 1/sqrt(64)
        if (kc == qt && col > row) s = -3.0e38f;  // causal mask (diag chunk)
        Ss[row][col] = s;
      }
    }
    __syncthreads();  // [B]

    // ---- online softmax: 256-thread two-level reduction ----
    {
      float mc = -3.0e38f;
#pragma unroll
      for (int j = 0; j < 16; ++j) mc = fmaxf(mc, Ss[prow][pseg * 16 + j]);
      pmax[prow][pseg] = mc;
    }
    __syncthreads();  // [C]
    if (tid < 64) {
      float mc = fmaxf(fmaxf(pmax[tid][0], pmax[tid][1]),
                       fmaxf(pmax[tid][2], pmax[tid][3]));
      float mold = mrow[tid];
      float mnew = fmaxf(mold, mc);
      arow[tid] = __expf(mold - mnew);
      mbc[tid]  = mnew;
      mrow[tid] = mnew;
    }
    __syncthreads();  // [D]
    {
      float mnew = mbc[prow], l = 0.0f;
#pragma unroll
      for (int j = 0; j < 16; ++j) {
        int c = pseg * 16 + j;
        float p = __expf(Ss[prow][c] - mnew);
        Ps[prow][c] = (__bf16)p;
        l += p;
      }
      psum[prow][pseg] = l;
    }
    __syncthreads();  // [E] Ps complete
    if (tid < 64)
      lrow[tid] = lrow[tid] * arow[tid] +
                  (psum[tid][0] + psum[tid][1]) + (psum[tid][2] + psum[tid][3]);

    // ---- rescale O, then O += P * V ----
#pragma unroll
    for (int t = 0; t < 2; ++t)
#pragma unroll
      for (int r = 0; r < 8; ++r)
        oacc[t][r] *= arow[rg * 16 + r + hi * 8];
#pragma unroll
    for (int step = 0; step < 2; ++step) {
      v16bf a = frag_a(&Ps[rg * 16 + ln][step * 32], hi);
#pragma unroll
      for (int t = 0; t < 2; ++t) {
        v16bf bb = frag_b(&Vs[(ct0 + t) * 16 + ln][step * 32], hi);
        oacc[t] = __builtin_amdgcn_wmma_f32_16x16x32_bf16(
            false, a, false, bb, (short)0, oacc[t], false, false);
      }
    }
    __syncthreads();  // [F] protect tiles for next chunk; publish lrow
  }

  __bf16* Obase = Og + ((size_t)(b * T_ + q0)) * D_ + h * HD_;
#pragma unroll
  for (int t = 0; t < 2; ++t) {
    int otile = ct0 + t;
#pragma unroll
    for (int r = 0; r < 8; ++r) {
      int row = rg * 16 + r + hi * 8;
      float v = oacc[t][r] / lrow[row];
      Obase[(size_t)row * D_ + otile * 16 + ln] = (__bf16)v;
    }
  }
}

// ---------------------------------------------------------------------------
// Host launcher. Inputs: x, mask(unused), Wq, Wk, Wv, Wo, bo.
// Workspace (88 MB): xb[0,16M) | WqT/WkT/WvT/WoT bf16 @16/18/20/22M |
// Q @24M | K @40M | V(transposed) @56M | attn-out @72M.
// ---------------------------------------------------------------------------
extern "C" void kernel_launch(void* const* d_in, const int* in_sizes, int n_in,
                              void* d_out, int out_size, void* d_ws,
                              size_t ws_size, hipStream_t stream) {
  const float* x  = (const float*)d_in[0];
  const float* Wq = (const float*)d_in[2];
  const float* Wk = (const float*)d_in[3];
  const float* Wv = (const float*)d_in[4];
  const float* Wo = (const float*)d_in[5];
  const float* bo = (const float*)d_in[6];
  float* out = (float*)d_out;

  const size_t MB = 1u << 20;
  char* ws = (char*)d_ws;
  __bf16* xb  = (__bf16*)(ws + 0);
  __bf16* wqt = (__bf16*)(ws + 16 * MB);
  __bf16* wkt = (__bf16*)(ws + 18 * MB);
  __bf16* wvt = (__bf16*)(ws + 20 * MB);
  __bf16* wot = (__bf16*)(ws + 22 * MB);
  __bf16* qb  = (__bf16*)(ws + 24 * MB);
  __bf16* kb  = (__bf16*)(ws + 40 * MB);
  __bf16* vb  = (__bf16*)(ws + 56 * MB);
  __bf16* ab  = (__bf16*)(ws + 72 * MB);

  // 1) fp32 -> bf16: x linear; weights transposed to [N][K]
  {
    int nX4 = (M_ * D_) / 4;
    f32_to_bf16_kernel<<<(nX4 + 255) / 256, 256, 0, stream>>>(x, xb, nX4);
    dim3 gt(D_ / 32, D_ / 32), bt(32, 8);
    transpose_f32_to_bf16_kernel<<<gt, bt, 0, stream>>>(Wq, wqt, D_, D_);
    transpose_f32_to_bf16_kernel<<<gt, bt, 0, stream>>>(Wk, wkt, D_, D_);
    transpose_f32_to_bf16_kernel<<<gt, bt, 0, stream>>>(Wv, wvt, D_, D_);
    transpose_f32_to_bf16_kernel<<<gt, bt, 0, stream>>>(Wo, wot, D_, D_);
  }

  // 2) Q/K/V projections (V written in [b][h][hd][T] layout)
  dim3 gproj(D_ / 128, M_ / 64);  // (8, 128)
  gemm_bf16_kernel<0><<<gproj, 256, 0, stream>>>(
      xb, wqt, qb, nullptr, nullptr, M_, D_, D_);
  gemm_bf16_kernel<0><<<gproj, 256, 0, stream>>>(
      xb, wkt, kb, nullptr, nullptr, M_, D_, D_);
  gemm_bf16_kernel<1><<<gproj, 256, 0, stream>>>(
      xb, wvt, vb, nullptr, nullptr, M_, D_, D_);

  // 3) fused causal flash attention: grid = B*H*(T/64) = 2048
  attn_kernel<<<B_ * H_ * (T_ / 64), 256, 0, stream>>>(qb, kb, vb, ab);

  // 4) output projection + bias -> f32
  gemm_bf16_kernel<2><<<gproj, 256, 0, stream>>>(
      ab, wot, nullptr, out, bo, M_, D_, D_);
}
